// CausalSelfAttention_42090679501213
// MI455X (gfx1250) — matrix operand order repaired and checked by reading, not compiled, not run on hardware
//
#include <hip/hip_runtime.h>

typedef __bf16 bf16;
typedef __attribute__((ext_vector_type(16))) bf16  v16bf;
typedef __attribute__((ext_vector_type(4)))  bf16  v4bf;
typedef __attribute__((ext_vector_type(8)))  float v8f;

#define D_MODEL 2048
#define N_HEADS 16
#define D_HEAD  128
#define BB      2
#define TT      2048
#define NTOK    (BB*TT)          // 4096
#define F_QKV   (3*D_MODEL)      // 6144

#define USE_ASYNC 1

static __device__ __forceinline__ bf16 f2bf(float f) { return (bf16)f; }

// ---- CDNA5 async global->LDS copy (ASYNCcnt-tracked, no VGPR round-trip) ----
static __device__ __forceinline__ void async_copy16(void* lds, const void* g) {
#if USE_ASYNC
    unsigned int l = (unsigned int)(size_t)lds;      // flat LDS aperture: low 32 bits = LDS byte addr
    asm volatile("global_load_async_to_lds_b128 %0, %1, off"
                 :: "v"(l), "v"(g) : "memory");
#else
    *(uint4*)lds = *(const uint4*)g;
#endif
}
static __device__ __forceinline__ void async_wait() {
#if USE_ASYNC
    asm volatile("s_wait_asynccnt 0x0" ::: "memory");
#endif
}
// wave-private LDS store->load ordering (LDS is in-order per wave; this also
// blocks compiler reordering)
static __device__ __forceinline__ void lds_fence() {
    asm volatile("s_wait_dscnt 0x0" ::: "memory");
}

// ---- vectorized fragment loaders (layouts per CDNA5 ISA 7.12.2) ----
// A-fragment 16x32 (MxK): elements 0..7 <- k=hi*8+0..7 ; 8..15 <- k=16+hi*8+0..7
static __device__ __forceinline__ v16bf load_a_frag(const bf16* __restrict__ p, int ld, int lane) {
    int m = lane & 15, hi = lane >> 4;
    const bf16* q = p + (size_t)m * ld + hi * 8;
    v16bf a;
    *(uint4*)&a       = *(const uint4*)(q);
    *((uint4*)&a + 1) = *(const uint4*)(q + 16);
    return a;
}
// B-fragment 32x16 (KxN), B[k][n] = rows[n][k]: elements 0..7 <- k=hi*8+0..7 ; 8..15 <- k=16+hi*8+0..7
static __device__ __forceinline__ v16bf load_b_frag(const bf16* __restrict__ p, int ld, int lane) {
    int n = lane & 15, hi = lane >> 4;
    const bf16* q = p + (size_t)n * ld + hi * 8;
    v16bf b;
    *(uint4*)&b       = *(const uint4*)(q);
    *((uint4*)&b + 1) = *(const uint4*)(q + 16);
    return b;
}
static __device__ __forceinline__ v8f wmma_bf16(v16bf a, v16bf b, v8f c) {
    return __builtin_amdgcn_wmma_f32_16x16x32_bf16(false, a, false, b, (short)0, c, false, false);
}

// ============ prep: f32 -> bf16 bulk convert ============
__global__ __launch_bounds__(256) void k_cvt_bf16(const float4* __restrict__ in,
                                                  v4bf* __restrict__ out) {
    int i = blockIdx.x * blockDim.x + threadIdx.x;
    float4 v = in[i];
    v4bf o;
    o[0] = f2bf(v.x); o[1] = f2bf(v.y); o[2] = f2bf(v.z); o[3] = f2bf(v.w);
    out[i] = o;
}

// ============ GEMM: C[M x N] = A[M x 2048] * W[N x 2048]^T ============
// 128x128 block tile, 8 waves (4m x 2n) of 32x64; A/B K-slices double-buffered in
// LDS (stride 40 = conflict-free) via async global->LDS copies.
#define GLD 40
template <typename OutT>
__global__ __launch_bounds__(256) void k_gemm(const bf16* __restrict__ A,
                                              const bf16* __restrict__ W,
                                              OutT* __restrict__ C, int N) {
    const int K = D_MODEL;
    __shared__ bf16 sA[2][128 * GLD];
    __shared__ bf16 sB[2][128 * GLD];

    int ntb = N >> 7;
    int mt = blockIdx.x / ntb, nt = blockIdx.x % ntb;
    int tid = threadIdx.x;
    int lane = tid & 31;
    int w = tid >> 5;
    int wm = w >> 1, wn = w & 1;                        // 4 x 2 wave grid

    const bf16* Ap = A + (size_t)(mt * 128) * K;
    const bf16* Wp = W + (size_t)(nt * 128) * K;

    v8f acc[2][4];
#pragma unroll
    for (int i = 0; i < 2; ++i)
#pragma unroll
        for (int j = 0; j < 4; ++j) acc[i][j] = (v8f){};

    auto stage = [&](int buf, int k0) {
#pragma unroll
        for (int i = 0; i < 2; ++i) {
            int c = tid + i * 256;                      // 512 16B chunks per tile
            int row = c >> 2, seg = c & 3;
            async_copy16(&sA[buf][row * GLD + seg * 8], Ap + (size_t)row * K + k0 + seg * 8);
            async_copy16(&sB[buf][row * GLD + seg * 8], Wp + (size_t)row * K + k0 + seg * 8);
        }
    };

    const int NK = K / 32;
    stage(0, 0);
    async_wait();
    __syncthreads();

    for (int ks = 0; ks < NK; ++ks) {
        int buf = ks & 1;
        if (ks + 1 < NK) stage(buf ^ 1, (ks + 1) * 32);

        const bf16* At = &sA[buf][0];
        const bf16* Bt = &sB[buf][0];
        v16bf a0 = load_a_frag(At + (wm * 32) * GLD, GLD, lane);
        v16bf a1 = load_a_frag(At + (wm * 32 + 16) * GLD, GLD, lane);
#pragma unroll
        for (int j = 0; j < 4; ++j) {
            v16bf b = load_b_frag(Bt + (wn * 64 + j * 16) * GLD, GLD, lane);
            acc[0][j] = wmma_bf16(a0, b, acc[0][j]);
            acc[1][j] = wmma_bf16(a1, b, acc[1][j]);
        }
        async_wait();
        __syncthreads();
    }

    int n = lane & 15, hi = lane >> 4;
#pragma unroll
    for (int i = 0; i < 2; ++i) {
#pragma unroll
        for (int j = 0; j < 4; ++j) {
            OutT* Cp = C + (size_t)(mt * 128 + wm * 32 + i * 16) * N + nt * 128 + wn * 64 + j * 16;
#pragma unroll
            for (int r = 0; r < 8; ++r)
                Cp[(size_t)(hi * 8 + r) * N + n] = (OutT)acc[i][j][r];
        }
    }
}

// ============ RoPE + gain + scale for q,k ============
__global__ __launch_bounds__(256) void k_rope(const bf16* __restrict__ qkv,
                                              const float* __restrict__ gain,
                                              const float* __restrict__ cosT,
                                              const float* __restrict__ sinT,
                                              bf16* __restrict__ qb,
                                              bf16* __restrict__ kb) {
    int idx = blockIdx.x * blockDim.x + threadIdx.x;
    int d2 = idx & 63;
    int h  = (idx >> 6) & 15;
    int t  = (idx >> 10) & 2047;
    int b  = idx >> 21;
    const bf16* base = qkv + (size_t)(b * TT + t) * F_QKV + h * D_HEAD;
    float q1 = (float)base[d2],           q2 = (float)base[64 + d2];
    float k1 = (float)base[D_MODEL + d2], k2 = (float)base[D_MODEL + 64 + d2];
    float c = cosT[t * 64 + d2], s = sinT[t * 64 + d2];
    float g = gain[h] * 0.08838834764831845f;
    size_t o = ((size_t)((b * N_HEADS + h) * TT + t)) * D_HEAD;
    qb[o + d2]      = f2bf((q1 * c - q2 * s) * g);
    qb[o + 64 + d2] = f2bf((q2 * c + q1 * s) * g);
    kb[o + d2]      = f2bf(k1 * c - k2 * s);
    kb[o + 64 + d2] = f2bf(k2 * c + k1 * s);
}

// ============ V transpose: qkv v-slice [B,T,H,D] -> vt [B,H,D,T] ============
__global__ __launch_bounds__(256) void k_vt(const bf16* __restrict__ qkv,
                                            bf16* __restrict__ vt) {
    __shared__ bf16 tile[32][33];
    int bh = blockIdx.z;
    int b = bh >> 4, h = bh & 15;
    int t0 = blockIdx.x * 32, d0 = blockIdx.y * 32;
    int tx = threadIdx.x, ty = threadIdx.y;
    const bf16* src = qkv + (size_t)(b * TT) * F_QKV + 2 * D_MODEL + h * D_HEAD;
#pragma unroll
    for (int i = 0; i < 32; i += 8)
        tile[ty + i][tx] = src[(size_t)(t0 + ty + i) * F_QKV + d0 + tx];
    __syncthreads();
    bf16* dst = vt + (size_t)bh * D_HEAD * TT;
#pragma unroll
    for (int i = 0; i < 32; i += 8)
        dst[(size_t)(d0 + ty + i) * TT + t0 + tx] = tile[tx][ty + i];
}

// ============ flash attention: 4 waves / block, shared K,V tiles in LDS ============
// sK: [32 keys][136] (pad), sV: [128 d][40] (pad), double buffered; async staged.
#define KLD 136
#define VLD 40
#define PLD 40
__global__ __launch_bounds__(128) void k_attn(const bf16* __restrict__ Q,
                                              const bf16* __restrict__ K,
                                              const bf16* __restrict__ Vt,
                                              bf16* __restrict__ Y) {
    __shared__ bf16 sK[2][32 * KLD];
    __shared__ bf16 sV[2][128 * VLD];
    __shared__ bf16 sP[4][16 * PLD];

    int tid = threadIdx.x;
    int lane = tid & 31;
    int w = tid >> 5;                                  // wave 0..3
    int qt = blockIdx.x * 4 + w;                       // q-tile per wave
    int bh = blockIdx.y;
    int b = bh >> 4, h = bh & 15;
    int m = lane & 15, hi = lane >> 4;

    const bf16* Qb = Q  + ((size_t)bh * TT + qt * 16) * D_HEAD;
    const bf16* Kb = K  + (size_t)bh * TT * D_HEAD;
    const bf16* Vb = Vt + (size_t)bh * D_HEAD * TT;    // [d][t]

    v16bf aq[4];
#pragma unroll
    for (int c = 0; c < 4; ++c)
        aq[c] = load_a_frag(Qb + c * 32, D_HEAD, lane);

    v8f o[8];
#pragma unroll
    for (int dt = 0; dt < 8; ++dt) o[dt] = (v8f){};
    float mrow[8], lrow[8];
#pragma unroll
    for (int r = 0; r < 8; ++r) { mrow[r] = -3.0e38f; lrow[r] = 0.0f; }

    auto stageKV = [&](int buf, int key0) {
#pragma unroll
        for (int i = 0; i < 4; ++i) {
            int c = tid + i * 128;                     // 512 chunks each
            int krow = c >> 4, kseg = c & 15;          // K tile: 32 rows x 256B
            async_copy16(&sK[buf][krow * KLD + kseg * 8],
                         Kb + (size_t)(key0 + krow) * D_HEAD + kseg * 8);
            int vrow = c >> 2, vseg = c & 3;           // V tile: 128 rows x 64B
            async_copy16(&sV[buf][vrow * VLD + vseg * 8],
                         Vb + (size_t)vrow * TT + key0 + vseg * 8);
        }
    };

    int my_nkb  = ((qt * 16 + 15) >> 5) + 1;
    int nkb_max = blockIdx.x * 2 + 2;                  // deepest wave's count

    stageKV(0, 0);
    async_wait();
    __syncthreads();

    for (int j = 0; j < nkb_max; ++j) {
        int buf = j & 1;
        if (j + 1 < nkb_max) stageKV(buf ^ 1, (j + 1) * 32);

        if (j < my_nkb) {                              // uniform per-wave branch
            int key0 = j * 32;
            v8f s0 = {}, s1 = {};
#pragma unroll
            for (int c = 0; c < 4; ++c) {
                v16bf bk0 = load_b_frag(&sK[buf][c * 32], KLD, lane);
                s0 = wmma_bf16(aq[c], bk0, s0);
                v16bf bk1 = load_b_frag(&sK[buf][16 * KLD + c * 32], KLD, lane);
                s1 = wmma_bf16(aq[c], bk1, s1);
            }
            int n0 = key0 + m, n1 = n0 + 16;
            float alpha[8];
#pragma unroll
            for (int r = 0; r < 8; ++r) {
                int qi = qt * 16 + hi * 8 + r;
                float x0 = (n0 <= qi) ? s0[r] : -3.0e38f;
                float x1 = (n1 <= qi) ? s1[r] : -3.0e38f;
                float mx = fmaxf(x0, x1);
#pragma unroll
                for (int off = 1; off < 16; off <<= 1)
                    mx = fmaxf(mx, __shfl_xor(mx, off, 32));
                float mn = fmaxf(mrow[r], mx);
                float p0 = __expf(x0 - mn);
                float p1 = __expf(x1 - mn);
                float a  = __expf(mrow[r] - mn);
                float rs = p0 + p1;
#pragma unroll
                for (int off = 1; off < 16; off <<= 1)
                    rs += __shfl_xor(rs, off, 32);
                lrow[r] = lrow[r] * a + rs;
                mrow[r] = mn;
                alpha[r] = a;
                s0[r] = p0;
                s1[r] = p1;
            }
            bf16* myP = &sP[w][0];
#pragma unroll
            for (int r = 0; r < 8; ++r) {
                myP[(hi * 8 + r) * PLD + m]      = f2bf(s0[r]);
                myP[(hi * 8 + r) * PLD + 16 + m] = f2bf(s1[r]);
            }
            lds_fence();                               // wave-private store->load ordering
            v16bf ap;
            *(uint4*)&ap       = *(const uint4*)(myP + m * PLD + hi * 8);
            *((uint4*)&ap + 1) = *(const uint4*)(myP + m * PLD + 16 + hi * 8);
#pragma unroll
            for (int dt = 0; dt < 8; ++dt) {
                v16bf bv = load_b_frag(&sV[buf][(dt * 16) * VLD], VLD, lane);
#pragma unroll
                for (int r = 0; r < 8; ++r) o[dt][r] *= alpha[r];
                o[dt] = wmma_bf16(ap, bv, o[dt]);
            }
        }
        async_wait();
        __syncthreads();
    }
#pragma unroll
    for (int dt = 0; dt < 8; ++dt) {
#pragma unroll
        for (int r = 0; r < 8; ++r) {
            int t = qt * 16 + hi * 8 + r;
            Y[((size_t)(b * TT + t)) * D_MODEL + h * D_HEAD + dt * 16 + m] =
                f2bf(o[dt][r] / lrow[r]);
        }
    }
}

extern "C" void kernel_launch(void* const* d_in, const int* in_sizes, int n_in,
                              void* d_out, int out_size, void* d_ws, size_t ws_size,
                              hipStream_t stream) {
    const float* x     = (const float*)d_in[0];
    const float* Wqkv  = (const float*)d_in[1];
    const float* Wproj = (const float*)d_in[2];
    const float* qgain = (const float*)d_in[3];
    const float* rcos  = (const float*)d_in[4];
    const float* rsin  = (const float*)d_in[5];
    float* out = (float*)d_out;

    char* ws = (char*)d_ws;
    bf16* xb     = (bf16*)(ws + 0);
    bf16* wqkvb  = (bf16*)(ws + 16777216);
    bf16* wprojb = (bf16*)(ws + 41943040);
    bf16* qkvb   = (bf16*)(ws + 50331648);              // reused as yb
    bf16* qb     = (bf16*)(ws + 100663296);
    bf16* kb     = (bf16*)(ws + 117440512);
    bf16* vt     = (bf16*)(ws + 134217728);
    bf16* yb     = qkvb;

    k_cvt_bf16<<<dim3((NTOK * D_MODEL / 4) / 256), 256, 0, stream>>>((const float4*)x, (v4bf*)xb);
    k_cvt_bf16<<<dim3((F_QKV * D_MODEL / 4) / 256), 256, 0, stream>>>((const float4*)Wqkv, (v4bf*)wqkvb);
    k_cvt_bf16<<<dim3((D_MODEL * D_MODEL / 4) / 256), 256, 0, stream>>>((const float4*)Wproj, (v4bf*)wprojb);

    k_gemm<bf16><<<dim3((NTOK / 128) * (F_QKV / 128)), 256, 0, stream>>>(xb, wqkvb, qkvb, F_QKV);
    k_rope<<<dim3((BB * TT * N_HEADS * 64) / 256), 256, 0, stream>>>(qkvb, qgain, rcos, rsin, qb, kb);
    k_vt<<<dim3(TT / 32, D_HEAD / 32, BB * N_HEADS), dim3(32, 8), 0, stream>>>(qkvb, vt);
    k_attn<<<dim3(TT / 16 / 4, BB * N_HEADS), 128, 0, stream>>>(qb, kb, vt, yb);
    k_gemm<float><<<dim3((NTOK / 128) * (D_MODEL / 128)), 256, 0, stream>>>(yb, wprojb, out, D_MODEL);

    (void)in_sizes; (void)n_in; (void)out_size; (void)ws_size;
}
